// SelfAttentionLayer_34428457845227
// MI455X (gfx1250) — matrix operand order repaired and checked by reading, not compiled, not run on hardware
//
#include <hip/hip_runtime.h>
#include <hip/hip_bf16.h>

// ---------------------------------------------------------------------------
// Self-attention on gfx1250 (MI455X). bf16 WMMA everywhere; double-buffered
// GLOBAL_LOAD_ASYNC_TO_LDS_B128 staging overlaps DMA with the WMMA stream.
//   B=2, S=2048, D=1024, H=16, Dh=64.
// Pipeline:
//   xprep : x f32 -> bf16                                   (8MB)
//   wprep : W{q,k,v,o} f32 [K][N] -> bf16 transposed [N][K] (4 x 2MB)
//   qkv   : xb @ Wt{q,k,v} + b -> Q,K [B,H,S,64], V^T [B,H,64,S] bf16
//   attn  : flash-style online softmax, WMMA scores + ctx
//   out   : ctx @ Wt_o + bo -> f32
// Workspace: xb(8) + Wt(8) + Q/K/Vt(24) + ctx(8) = 48MB.
// ---------------------------------------------------------------------------

typedef __bf16 bf16_t;
typedef __attribute__((ext_vector_type(16))) __bf16 v16bf;
typedef __attribute__((ext_vector_type(8)))  __bf16 bf8;
typedef __attribute__((ext_vector_type(4)))  __bf16 bf4;
typedef __attribute__((ext_vector_type(8)))  float  v8f;
typedef __attribute__((ext_vector_type(4)))  float  f4v;

#define D_MODEL 1024
#define S_LEN   2048
#define NHEAD   16
#define DHEAD   64
#define LDS_STRIDE 40            // bf16 elems; 80B rows, 16B-aligned fragments
#define TILE_ELEMS (128 * LDS_STRIDE)

__device__ __forceinline__ bf16_t tobf(float f) {
  unsigned u = __builtin_bit_cast(unsigned, f);
  u += 0x7FFFu + ((u >> 16) & 1u);              // round-to-nearest-even
  unsigned short h = (unsigned short)(u >> 16);
  return __builtin_bit_cast(bf16_t, h);
}

union V16U { v16bf v; bf8 h[2]; };
__device__ __forceinline__ v16bf combine(bf8 lo, bf8 hi) {
  V16U u; u.h[0] = lo; u.h[1] = hi; return u.v;
}

__device__ __forceinline__ v8f v8zero() {
  v8f z = {0.f,0.f,0.f,0.f,0.f,0.f,0.f,0.f}; return z;
}

__device__ __forceinline__ v8f wmma_bf16(v16bf a, v16bf b, v8f c) {
  return __builtin_amdgcn_wmma_f32_16x16x32_bf16(false, a, false, b,
                                                 (short)0, c, false, false);
}

// Async global->LDS 16-byte copy (GLOBAL_LOAD_ASYNC_TO_LDS_B128, ASYNCcnt).
// ldsAddr = wave-relative LDS byte address (low 32 bits of the generic
// shared-aperture pointer).
__device__ __forceinline__ void async_copy_b128(unsigned ldsAddr,
                                                unsigned long long gaddr) {
  asm volatile("global_load_async_to_lds_b128 %0, %1, off"
               :: "v"(ldsAddr), "v"(gaddr) : "memory");
}
__device__ __forceinline__ void wait_async0() {
  asm volatile("s_wait_asynccnt 0x0" ::: "memory");
}

// ---------------------------------------------------------------------------
// Prep 1: x f32 -> bf16. grid 1024 x 256, 4 float4 per thread (4M elements).
// ---------------------------------------------------------------------------
__global__ __launch_bounds__(256) void xprep_kernel(const float* __restrict__ x,
                                                    bf16_t* __restrict__ xb) {
#pragma unroll
  for (int i = 0; i < 4; ++i) {
    size_t c = (size_t)blockIdx.x * 1024 + threadIdx.x + i * 256;  // float4 idx
    f4v f = *(const f4v*)(x + c * 4);
    bf4 b; b[0]=tobf(f[0]); b[1]=tobf(f[1]); b[2]=tobf(f[2]); b[3]=tobf(f[3]);
    *(bf4*)(xb + c * 4) = b;
  }
}

// ---------------------------------------------------------------------------
// Prep 2: W [K][N] f32 -> Wt [N][K] bf16 (convert + transpose via LDS tile).
// grid (16,16,4) x 256; 64x64 tile per block.
// ---------------------------------------------------------------------------
__global__ __launch_bounds__(256) void wprep_kernel(
    const float* __restrict__ Wq, const float* __restrict__ Wk,
    const float* __restrict__ Wv, const float* __restrict__ Wo,
    bf16_t* __restrict__ Wtq, bf16_t* __restrict__ Wtk,
    bf16_t* __restrict__ Wtv, bf16_t* __restrict__ Wto) {
  __shared__ __align__(16) bf16_t T[64 * 72];   // [n][k], stride 72 (144B)
  const int z = blockIdx.z;
  const float* W = (z == 0) ? Wq : (z == 1) ? Wk : (z == 2) ? Wv : Wo;
  bf16_t*     Wt = (z == 0) ? Wtq : (z == 1) ? Wtk : (z == 2) ? Wtv : Wto;
  const int k0 = blockIdx.y * 64, n0 = blockIdx.x * 64;
  const int tid = threadIdx.x;
#pragma unroll
  for (int i = 0; i < 4; ++i) {
    int idx = tid + i * 256;                    // 0..1023 float4 chunks
    int kr = idx >> 4, n4 = (idx & 15) * 4;
    f4v f = *(const f4v*)(W + (size_t)(k0 + kr) * D_MODEL + n0 + n4);
    T[(n4 + 0) * 72 + kr] = tobf(f[0]);
    T[(n4 + 1) * 72 + kr] = tobf(f[1]);
    T[(n4 + 2) * 72 + kr] = tobf(f[2]);
    T[(n4 + 3) * 72 + kr] = tobf(f[3]);
  }
  __syncthreads();
#pragma unroll
  for (int i = 0; i < 2; ++i) {
    int idx = tid + i * 256;                    // 0..511 bf8 chunks
    int nr = idx >> 3, c8 = (idx & 7) * 8;
    bf8 b = *(const bf8*)(T + nr * 72 + c8);
    *(bf8*)(Wt + (size_t)(n0 + nr) * D_MODEL + k0 + c8) = b;
  }
}

// ---------------------------------------------------------------------------
// One compute step of the GEMM: fragments from one LDS buffer + 8 WMMAs.
// ---------------------------------------------------------------------------
__device__ __forceinline__ void gemm_compute(const bf16_t* Al, const bf16_t* Bl,
                                             int wm, int wn, int lh, int l15,
                                             v8f acc[2][4]) {
  // A fragments (lane M=l15, 16 contiguous K by lh, same order as B)
  v16bf afr[2];
#pragma unroll
  for (int sm = 0; sm < 2; ++sm) {
    const bf16_t* ap = Al + (wm + sm * 16 + l15) * LDS_STRIDE;
    afr[sm] = combine(*(const bf8*)(ap + 16 * lh),
                      *(const bf8*)(ap + 16 * lh + 8));
  }
  // B fragments (lane N=l15, 16 contiguous K by lh) + 8 WMMAs
#pragma unroll
  for (int t = 0; t < 4; ++t) {
    const bf16_t* bp = Bl + (wn + t * 16 + l15) * LDS_STRIDE + 16 * lh;
    v16bf bfr = combine(*(const bf8*)bp, *(const bf8*)(bp + 8));
#pragma unroll
    for (int sm = 0; sm < 2; ++sm)
      acc[sm][t] = wmma_bf16(afr[sm], bfr, acc[sm][t]);
  }
}

// ---------------------------------------------------------------------------
// GEMM mainloop: C[128x128] += A[128xK] * Bt[128xK]^T, both bf16, K=1024.
// Double-buffered async-to-LDS staging: while buffer `cur` feeds the WMMAs,
// GLOBAL_LOAD_ASYNC_TO_LDS_B128s fill the other buffer; s_wait_asynccnt 0 +
// split barrier closes each half-step. k unrolled by 64 so the buffer index
// is a compile-time constant.
// Block 256 = 8 waves; wave grid 4(M) x 2(N); each wave 32x64 (8 WMMA accs).
// ---------------------------------------------------------------------------
__device__ __forceinline__ void gemm_mainloop_bf(
    const bf16_t* __restrict__ A, const bf16_t* __restrict__ Bt,
    bf16_t (*Alds)[TILE_ELEMS], bf16_t (*Blds)[TILE_ELEMS],
    int m0, int n0, v8f acc[2][4]) {
  const int tid = threadIdx.x;
  const int wave = tid >> 5, lane = tid & 31;
  const int lh = lane >> 4, l15 = lane & 15;
  const int wm = (wave & 3) * 32, wn = (wave >> 2) * 64;

#pragma unroll
  for (int sm = 0; sm < 2; ++sm)
#pragma unroll
    for (int t = 0; t < 4; ++t) acc[sm][t] = v8zero();

  // Per-thread staging slots (loop-invariant LDS dests, global bases w/o k0).
  unsigned ldsA[2][2], ldsB[2][2];   // [buffer][slot]
  const bf16_t* gA[2];
  const bf16_t* gB[2];
#pragma unroll
  for (int i = 0; i < 2; ++i) {
    int idx = tid + i * 256;                    // 0..511 bf8 chunks (128x32)
    int row = idx >> 2, c8 = (idx & 3) * 8;
#pragma unroll
    for (int bb = 0; bb < 2; ++bb) {
      ldsA[bb][i] = (unsigned)(uintptr_t)(&Alds[bb][row * LDS_STRIDE + c8]);
      ldsB[bb][i] = (unsigned)(uintptr_t)(&Blds[bb][row * LDS_STRIDE + c8]);
    }
    gA[i] = A + (size_t)(m0 + row) * D_MODEL + c8;
    gB[i] = Bt + (size_t)(n0 + row) * D_MODEL + c8;
  }

  // Prologue: fill buffer 0 with the k0=0 tiles.
#pragma unroll
  for (int i = 0; i < 2; ++i) {
    async_copy_b128(ldsA[0][i], (unsigned long long)(gA[i]));
    async_copy_b128(ldsB[0][i], (unsigned long long)(gB[i]));
  }
  wait_async0();
  __syncthreads();

  for (int k0 = 0; k0 < D_MODEL; k0 += 64) {
    // ---- half-step 1: prefetch buf1 @ k0+32, compute buf0 @ k0 ----
#pragma unroll
    for (int i = 0; i < 2; ++i) {
      async_copy_b128(ldsA[1][i], (unsigned long long)(gA[i] + k0 + 32));
      async_copy_b128(ldsB[1][i], (unsigned long long)(gB[i] + k0 + 32));
    }
    gemm_compute(Alds[0], Blds[0], wm, wn, lh, l15, acc);
    wait_async0();
    __syncthreads();

    // ---- half-step 2: prefetch buf0 @ k0+64 (if any), compute buf1 ----
    if (k0 + 64 < D_MODEL) {
#pragma unroll
      for (int i = 0; i < 2; ++i) {
        async_copy_b128(ldsA[0][i], (unsigned long long)(gA[i] + k0 + 64));
        async_copy_b128(ldsB[0][i], (unsigned long long)(gB[i] + k0 + 64));
      }
    }
    gemm_compute(Alds[1], Blds[1], wm, wn, lh, l15, acc);
    wait_async0();
    __syncthreads();
  }
}

// ---------------------------------------------------------------------------
// Kernel: QKV projections. grid = (8, 32, 3), block = 256.
// z=0 -> Q [B,H,S,64]; z=1 -> K [B,H,S,64]; z=2 -> V transposed [B,H,64,S].
// ---------------------------------------------------------------------------
__global__ __launch_bounds__(256) void qkv_kernel(
    const bf16_t* __restrict__ xb,
    const bf16_t* __restrict__ Wtq, const bf16_t* __restrict__ Wtk,
    const bf16_t* __restrict__ Wtv,
    const float* __restrict__ bq, const float* __restrict__ bk,
    const float* __restrict__ bv,
    bf16_t* __restrict__ Qb, bf16_t* __restrict__ Kb, bf16_t* __restrict__ Vt) {
  __shared__ __align__(16) bf16_t Alds[2][TILE_ELEMS];
  __shared__ __align__(16) bf16_t Blds[2][TILE_ELEMS];
  const int z = blockIdx.z;
  const bf16_t* Wt   = (z == 0) ? Wtq : (z == 1) ? Wtk : Wtv;
  const float*  bias = (z == 0) ? bq : (z == 1) ? bk : bv;
  const int m0 = blockIdx.y * 128, n0 = blockIdx.x * 128;

  v8f acc[2][4];
  gemm_mainloop_bf(xb, Wt, Alds, Blds, m0, n0, acc);

  const int tid = threadIdx.x, wave = tid >> 5, lane = tid & 31;
  const int lh = lane >> 4, l15 = lane & 15;
  const int mBase = m0 + (wave & 3) * 32;
  const int nBase = n0 + (wave >> 2) * 64;
#pragma unroll
  for (int sm = 0; sm < 2; ++sm) {
#pragma unroll
    for (int t = 0; t < 4; ++t) {
      int n = nBase + t * 16 + l15;
      float bn = bias[n];
      int h = n >> 6, dc = n & 63;
#pragma unroll
      for (int r = 0; r < 8; ++r) {
        int m = mBase + sm * 16 + r + 8 * lh;   // C/D layout: lanes16-31 -> M+8
        int b = m >> 11, s = m & (S_LEN - 1);
        int bh = b * NHEAD + h;
        float v = acc[sm][t][r] + bn;
        if (z == 2)
          Vt[((size_t)bh * DHEAD + dc) * S_LEN + s] = tobf(v);
        else
          ((z == 0) ? Qb : Kb)[((size_t)bh * S_LEN + s) * DHEAD + dc] = tobf(v);
      }
    }
  }
}

// ---------------------------------------------------------------------------
// Kernel: flash attention. grid = (S/64=32, B*H=32), block = 128 (4 waves).
// Each wave owns 16 query rows; online softmax over 64-key tiles.
// ---------------------------------------------------------------------------
__global__ __launch_bounds__(128) void attn_kernel(
    const bf16_t* __restrict__ Qb, const bf16_t* __restrict__ Kb,
    const bf16_t* __restrict__ Vt, bf16_t* __restrict__ ctx) {
  __shared__ __align__(16) bf16_t Plds[4 * 16 * 64];  // per-wave P scratch
  const int tid = threadIdx.x, wave = tid >> 5, lane = tid & 31;
  const int lh = lane >> 4, l15 = lane & 15;
  const int bh = blockIdx.y;
  const int q0 = blockIdx.x * 64 + wave * 16;

  const bf16_t* Kbase = Kb + (size_t)bh * S_LEN * DHEAD;
  const bf16_t* Vbase = Vt + (size_t)bh * DHEAD * S_LEN;
  bf16_t* Pw = Plds + wave * 16 * 64;

  // Q A-fragments for both d-halves (K-steps d=0..31, 32..63), global b128s.
  v16bf qfr[2];
  {
    const bf16_t* ap = Qb + ((size_t)bh * S_LEN + q0 + l15) * DHEAD;
    qfr[0] = combine(*(const bf8*)(ap + 16 * lh),      *(const bf8*)(ap + 16 * lh + 8));
    qfr[1] = combine(*(const bf8*)(ap + 32 + 16 * lh), *(const bf8*)(ap + 40 + 16 * lh));
  }

  float mrow[8], lrow[8];
#pragma unroll
  for (int r = 0; r < 8; ++r) { mrow[r] = -1e30f; lrow[r] = 0.f; }
  v8f cacc[4];
#pragma unroll
  for (int t = 0; t < 4; ++t) cacc[t] = v8zero();

  const float scale = 0.125f;  // 1/sqrt(64)

  for (int kt = 0; kt < S_LEN; kt += 64) {
    // ---- scores S = Q K^T (B-fragment straight from K natural layout) ----
    v8f sacc[4];
#pragma unroll
    for (int t = 0; t < 4; ++t) {
      sacc[t] = v8zero();
      const bf16_t* kp = Kbase + (size_t)(kt + t * 16 + l15) * DHEAD;
      v16bf b0 = combine(*(const bf8*)(kp + 16 * lh),      *(const bf8*)(kp + 16 * lh + 8));
      v16bf b1 = combine(*(const bf8*)(kp + 32 + 16 * lh), *(const bf8*)(kp + 40 + 16 * lh));
      sacc[t] = wmma_bf16(qfr[0], b0, sacc[t]);
      sacc[t] = wmma_bf16(qfr[1], b1, sacc[t]);
    }
    // ---- online softmax (row stats across the 16-lane half of wave32) ----
    float alpha[8];
#pragma unroll
    for (int r = 0; r < 8; ++r) {
      float mx = -1e30f;
#pragma unroll
      for (int t = 0; t < 4; ++t) { sacc[t][r] *= scale; mx = fmaxf(mx, sacc[t][r]); }
#pragma unroll
      for (int off = 8; off >= 1; off >>= 1) mx = fmaxf(mx, __shfl_xor(mx, off, 32));
      float mn = fmaxf(mrow[r], mx);
      alpha[r] = __expf(mrow[r] - mn);
      mrow[r] = mn;
      float ps = 0.f;
#pragma unroll
      for (int t = 0; t < 4; ++t) {
        float p = __expf(sacc[t][r] - mn);
        ps += p;
        Pw[(r + 8 * lh) * 64 + t * 16 + l15] = tobf(p);  // C/D -> row-major
      }
#pragma unroll
      for (int off = 8; off >= 1; off >>= 1) ps += __shfl_xor(ps, off, 32);
      lrow[r] = alpha[r] * lrow[r] + ps;
    }
#pragma unroll
    for (int t = 0; t < 4; ++t)
#pragma unroll
      for (int r = 0; r < 8; ++r) cacc[t][r] *= alpha[r];

    // wave-private LDS store->load; make the ordering explicit
    asm volatile("s_wait_dscnt 0x0" ::: "memory");

    // ---- ctx += P @ V (V-fragment straight from transposed V layout) ----
#pragma unroll
    for (int j = 0; j < 2; ++j) {
      const bf16_t* pp = Pw + l15 * 64 + j * 32;
      v16bf pfr = combine(*(const bf8*)(pp + 16 * lh), *(const bf8*)(pp + 16 * lh + 8));
#pragma unroll
      for (int t = 0; t < 4; ++t) {
        const bf16_t* vp = Vbase + (size_t)(t * 16 + l15) * S_LEN + kt + j * 32;
        v16bf vfr = combine(*(const bf8*)(vp + 16 * lh), *(const bf8*)(vp + 16 * lh + 8));
        cacc[t] = wmma_bf16(pfr, vfr, cacc[t]);
      }
    }
  }

  // ---- normalize and write ctx as [B,S,D] bf16 ----
  const int b = bh >> 4, h = bh & 15;
#pragma unroll
  for (int t = 0; t < 4; ++t) {
#pragma unroll
    for (int r = 0; r < 8; ++r) {
      int m = q0 + r + 8 * lh;
      int dc = t * 16 + l15;
      float v = cacc[t][r] / lrow[r];
      ctx[((size_t)(b * S_LEN + m)) * D_MODEL + h * DHEAD + dc] = tobf(v);
    }
  }
}

// ---------------------------------------------------------------------------
// Kernel: out = ctx @ Wo + bo (fp32). grid = (8, 32), block = 256.
// ---------------------------------------------------------------------------
__global__ __launch_bounds__(256) void out_kernel(
    const bf16_t* __restrict__ ctx, const bf16_t* __restrict__ Wto,
    const float* __restrict__ bo, float* __restrict__ out) {
  __shared__ __align__(16) bf16_t Alds[2][TILE_ELEMS];
  __shared__ __align__(16) bf16_t Blds[2][TILE_ELEMS];
  const int m0 = blockIdx.y * 128, n0 = blockIdx.x * 128;

  v8f acc[2][4];
  gemm_mainloop_bf(ctx, Wto, Alds, Blds, m0, n0, acc);

  const int tid = threadIdx.x, wave = tid >> 5, lane = tid & 31;
  const int lh = lane >> 4, l15 = lane & 15;
  const int mBase = m0 + (wave & 3) * 32;
  const int nBase = n0 + (wave >> 2) * 64;
#pragma unroll
  for (int sm = 0; sm < 2; ++sm) {
#pragma unroll
    for (int t = 0; t < 4; ++t) {
      int n = nBase + t * 16 + l15;
      float bn = bo[n];
#pragma unroll
      for (int r = 0; r < 8; ++r) {
        int m = mBase + sm * 16 + r + 8 * lh;
        out[(size_t)m * D_MODEL + n] = acc[sm][t][r] + bn;
      }
    }
  }
}

// ---------------------------------------------------------------------------
extern "C" void kernel_launch(void* const* d_in, const int* in_sizes, int n_in,
                              void* d_out, int out_size, void* d_ws, size_t ws_size,
                              hipStream_t stream) {
  const float* x  = (const float*)d_in[0];
  const float* Wq = (const float*)d_in[1];
  const float* bq = (const float*)d_in[2];
  const float* Wk = (const float*)d_in[3];
  const float* bk = (const float*)d_in[4];
  const float* Wv = (const float*)d_in[5];
  const float* bv = (const float*)d_in[6];
  const float* Wo = (const float*)d_in[7];
  const float* bo = (const float*)d_in[8];
  float* out = (float*)d_out;

  char* ws = (char*)d_ws;
  const size_t MB = 1024 * 1024;
  bf16_t* xb  = (bf16_t*)(ws + 0 * MB);
  bf16_t* Wtq = (bf16_t*)(ws + 8 * MB);
  bf16_t* Wtk = (bf16_t*)(ws + 10 * MB);
  bf16_t* Wtv = (bf16_t*)(ws + 12 * MB);
  bf16_t* Wto = (bf16_t*)(ws + 14 * MB);
  bf16_t* Qb  = (bf16_t*)(ws + 16 * MB);
  bf16_t* Kb  = (bf16_t*)(ws + 24 * MB);
  bf16_t* Vt  = (bf16_t*)(ws + 32 * MB);
  bf16_t* ctx = (bf16_t*)(ws + 40 * MB);

  xprep_kernel<<<1024, 256, 0, stream>>>(x, xb);
  dim3 gW(16, 16, 4);
  wprep_kernel<<<gW, 256, 0, stream>>>(Wq, Wk, Wv, Wo, Wtq, Wtk, Wtv, Wto);

  dim3 gQKV(D_MODEL / 128, (2 * S_LEN) / 128, 3);
  qkv_kernel<<<gQKV, 256, 0, stream>>>(xb, Wtq, Wtk, Wtv, bq, bk, bv, Qb, Kb, Vt);

  dim3 gAtt(S_LEN / 64, 2 * NHEAD, 1);
  attn_kernel<<<gAtt, 128, 0, stream>>>(Qb, Kb, Vt, ctx);

  dim3 gOut(D_MODEL / 128, (2 * S_LEN) / 128, 1);
  out_kernel<<<gOut, 256, 0, stream>>>(ctx, Wto, bo, out);
}
